// Model_35175782154259
// MI455X (gfx1250) — matrix operand order repaired and checked by reading, not compiled, not run on hardware
//
#include <hip/hip_runtime.h>
#include <hip/hip_bf16.h>

// HGP-SL graph classifier, fully fused: one workgroup (256 thr = 8 wave32) per graph.
// All per-graph state lives in LDS; every GEMM uses v_wmma_f32_16x16x32_f16 with
// dual-tile accumulator interleaving to hide the WMMA->WMMA RAW hazard.

#define NB   1024
#define NN   128
#define FF   128
#define KP1  64
#define KP2  32

typedef __attribute__((ext_vector_type(16))) _Float16 v16h;
typedef __attribute__((ext_vector_type(8)))  float    v8f;

#define WMMA_F16(a, b, c) \
  __builtin_amdgcn_wmma_f32_16x16x32_f16(false, (a), false, (b), (short)0, (c), false, false)

// ---- WMMA fragment loaders (contiguous k order, wave32) --------------------

// A-matrix 16x32 f16: lane = M%16; lanes 0-15 hold K=k0..k0+15, lanes 16-31 hold K=k0+16..k0+31
template<int LD>
__device__ inline v16h wmma_load_a(const _Float16* s, int m0, int k0, int lane) {
  v16h a;
  const int m  = m0 + (lane & 15);
  const int kb = (lane & 16) ? 16 : 0;
#pragma unroll
  for (int e = 0; e < 16; ++e) {
    int k = k0 + kb + e;
    a[e] = s[m * LD + k];
  }
  return a;
}

// B-matrix 32x16 f16: lanes 0-15 hold K=k0..k0+15, lanes 16-31 hold K=k0+16..k0+31
template<int LD>
__device__ inline v16h wmma_load_b(const _Float16* s, int k0, int n0, int lane) {
  v16h b;
  const int n  = n0 + (lane & 15);
  const int kb = k0 + ((lane & 16) ? 16 : 0);
#pragma unroll
  for (int e = 0; e < 16; ++e) b[e] = s[(kb + e) * LD + n];
  return b;
}

// ---- Tiled GEMM: C_f16[MxN] = act(A_f16[MxK] @ B_f16[KxN] + bias) ----------
// Each wave owns tile pair (t, t+8): same tile-column (TN==8), adjacent tile-rows.
// The pair shares one B fragment and provides two independent WMMA chains.
template<int M, int N, int K, int LDA, int LDB, int LDC, bool BIAS, bool RELU>
__device__ void mm_f16(const _Float16* __restrict__ A, const _Float16* __restrict__ B,
                       _Float16* __restrict__ C, const float* __restrict__ bias,
                       int wave, int lane) {
  constexpr int TN    = N >> 4;
  constexpr int TILES = (M >> 4) * TN;
  static_assert(TN == 8 && (TILES % 16) == 0, "pairing requires TN==8, TILES%16==0");
  for (int t = wave; t < TILES; t += 16) {
    const int tm = t / TN, tn = t - tm * TN;      // pair covers rows tm*16 .. tm*16+31
    v8f c0 = {}, c1 = {};
#pragma unroll
    for (int k0 = 0; k0 < K; k0 += 32) {
      v16h b  = wmma_load_b<LDB>(B, k0, tn * 16, lane);
      v16h a0 = wmma_load_a<LDA>(A, tm * 16,      k0, lane);
      v16h a1 = wmma_load_a<LDA>(A, tm * 16 + 16, k0, lane);
      c0 = WMMA_F16(a0, b, c0);                   // independent chains -> no v_nop bubbles
      c1 = WMMA_F16(a1, b, c1);
    }
    const int n     = tn * 16 + (lane & 15);
    const int mbase = tm * 16 + ((lane & 16) ? 8 : 0);
    float bn = 0.0f;
    if (BIAS) bn = bias[n];
#pragma unroll
    for (int r = 0; r < 8; ++r) {
      float v0 = c0[r], v1 = c1[r];
      if (BIAS) { v0 += bn; v1 += bn; }
      if (RELU) { v0 = fmaxf(v0, 0.0f); v1 = fmaxf(v1, 0.0f); }
      C[(mbase + r) * LDC + n]        = (_Float16)v0;
      C[(mbase + 16 + r) * LDC + n]   = (_Float16)v1;
    }
  }
}

// ---- Info-score GEMM: score[m] += sum_n | H[m,n] - dinv[m]*(A@H)[m,n] | ----
// Same dual-tile pairing; lane-shuffle reduction then one ds_add_f32 per row-chunk.
template<int M, int N, int K, int LDA, int LDH>
__device__ void mm_score(const _Float16* __restrict__ A, const _Float16* __restrict__ H,
                         const float* __restrict__ dinv, float* __restrict__ score,
                         int wave, int lane) {
  constexpr int TN    = N >> 4;
  constexpr int TILES = (M >> 4) * TN;
  static_assert(TN == 8 && (TILES % 16) == 0, "pairing requires TN==8, TILES%16==0");
  for (int t = wave; t < TILES; t += 16) {
    const int tm = t / TN, tn = t - tm * TN;
    v8f c0 = {}, c1 = {};
#pragma unroll
    for (int k0 = 0; k0 < K; k0 += 32) {
      v16h b  = wmma_load_b<LDH>(H, k0, tn * 16, lane);
      v16h a0 = wmma_load_a<LDA>(A, tm * 16,      k0, lane);
      v16h a1 = wmma_load_a<LDA>(A, tm * 16 + 16, k0, lane);
      c0 = WMMA_F16(a0, b, c0);
      c1 = WMMA_F16(a1, b, c1);
    }
    const int n     = tn * 16 + (lane & 15);
    const int mbase = tm * 16 + ((lane & 16) ? 8 : 0);
#pragma unroll
    for (int half = 0; half < 2; ++half) {
      const v8f& c = half ? c1 : c0;
      const int mb = mbase + half * 16;
#pragma unroll
      for (int r = 0; r < 8; ++r) {
        const int m = mb + r;                     // same m for all 16 lanes of a half-wave
        float h = (float)H[m * LDH + n];
        float v = fabsf(h - dinv[m] * c[r]);
        v += __shfl_xor(v, 1);
        v += __shfl_xor(v, 2);
        v += __shfl_xor(v, 4);
        v += __shfl_xor(v, 8);
        if ((lane & 15) == 0) atomicAdd(&score[m], v);   // ds_add_f32
      }
    }
  }
}

// ---- Wave-parallel argmax with lowest-index tie-break (jax top_k order) ----
__device__ inline int wave_argmax(volatile float* score, int n, int lane) {
  float best = -1.0f;            // scores are L1 norms >= 0; -1 marks "taken"
  int   bi   = 0x7fffffff;
  for (int j = lane; j < n; j += 32) {
    float v = score[j];
    if (v > best || (v == best && j < bi)) { best = v; bi = j; }
  }
#pragma unroll
  for (int off = 16; off >= 1; off >>= 1) {
    float ob = __shfl_xor(best, off);
    int   oi = __shfl_xor(bi, off);
    if (ob > best || (ob == best && oi < bi)) { best = ob; bi = oi; }
  }
  return bi;                     // all lanes hold the winner
}

// ---------------------------------------------------------------------------
__global__ void __launch_bounds__(256)
hgpsl_fused_kernel(const float* __restrict__ x,   const float* __restrict__ adj,
                   const float* __restrict__ W1,  const float* __restrict__ b1,
                   const float* __restrict__ W2,  const float* __restrict__ b2,
                   const float* __restrict__ W3,  const float* __restrict__ b3,
                   const float* __restrict__ att1,const float* __restrict__ att2,
                   const float* __restrict__ l1w, const float* __restrict__ l1b,
                   const float* __restrict__ l2w, const float* __restrict__ l2b,
                   const float* __restrict__ l3w, const float* __restrict__ l3b,
                   float* __restrict__ out) {
  __shared__ _Float16 sA[NN * FF];        // 32 KB
  __shared__ _Float16 sB[NN * FF];        // 32 KB
  __shared__ _Float16 sT[NN * FF];        // 32 KB
  __shared__ float    sAk[KP1 * KP1];     // 16 KB  (learned adjacency a1)
  __shared__ _Float16 sXk[KP1 * FF];      // 16 KB  (pooled features, level 1)
  __shared__ _Float16 sXk2[KP2 * FF];     //  8 KB  (pooled features, level 2)
  __shared__ float    sAk2[KP2 * KP2];    //  4 KB  (learned adjacency a2)
  __shared__ float    sDinv[NN];
  __shared__ float    sScore[NN];
  __shared__ int      sIdx[KP1];
  __shared__ float    sGacc[2 * FF];
  __shared__ float    sVec[2 * FF];

  const int tid  = threadIdx.x;
  const int lane = tid & 31;
  const int wave = tid >> 5;
  const int bidx = blockIdx.x;
  const float* xg = x   + (size_t)bidx * NN * FF;
  const float* ag = adj + (size_t)bidx * NN * NN;

  // Pull W1 toward the caches while we stage adj/x (global_prefetch_b8).
#pragma unroll
  for (int o = 0; o < FF * FF; o += 256 * 16)
    __builtin_prefetch(W1 + o + tid * 16, 0, 1);

  // ===== GCN layer 1 ========================================================
  for (int i = tid; i < NN * NN; i += 256) {         // coalesced: adj -> sA, x -> sB (f16)
    sA[i] = (_Float16)ag[i];
    sB[i] = (_Float16)xg[i];
  }
  if (tid < NN) sScore[tid] = 0.f;
  if (tid < 2 * FF) sGacc[tid] = 0.f;
  __syncthreads();
  if (tid < NN) {                                    // deg(A+I) from LDS
    float s = 1.0f;
    for (int j = 0; j < NN; ++j) s += (float)sA[tid * NN + j];
    sDinv[tid] = rsqrtf(fmaxf(s, 1e-12f));
  }
  __syncthreads();
  for (int i = tid; i < NN * NN; i += 256) {         // in-place: A -> D^-1/2 (A+I) D^-1/2
    int r = i >> 7, c = i & 127;
    float av = (float)sA[i] + (r == c ? 1.0f : 0.0f);
    sA[i] = (_Float16)(sDinv[r] * av * sDinv[c]);
  }
  __syncthreads();
  mm_f16<NN, FF, NN, FF, FF, FF, false, false>(sA, sB, sT, nullptr, wave, lane); // T = An@X
  __syncthreads();
  for (int i = tid; i < FF * FF; i += 256) sA[i] = (_Float16)W1[i];
  __syncthreads();
  mm_f16<NN, FF, FF, FF, FF, FF, true, true>(sT, sA, sB, b1, wave, lane);        // h = relu(T@W1+b1)
#pragma unroll
  for (int o = 0; o < FF * FF; o += 256 * 16)
    __builtin_prefetch(W2 + o + tid * 16, 0, 1);
  __syncthreads();

  // ===== Pool 1: info score on raw adj ======================================
  for (int i = tid; i < NN * NN; i += 256) sT[i] = (_Float16)ag[i];              // raw adj f16
  __syncthreads();
  if (tid < NN) {                                    // dinv = 1/max(rowsum(A),1)
    float s = 0.f;
    for (int j = 0; j < NN; ++j) s += (float)sT[tid * NN + j];
    sDinv[tid] = 1.0f / fmaxf(s, 1.0f);
  }
  __syncthreads();
  mm_score<NN, FF, NN, NN, FF>(sT, sB, sDinv, sScore, wave, lane);
  __syncthreads();
  if (wave == 0) {                                   // parallel top-K1 (desc, low-index ties)
    for (int s = 0; s < KP1; ++s) {
      int bi = wave_argmax(sScore, NN, lane);
      if (lane == 0) { sIdx[s] = bi; ((volatile float*)sScore)[bi] = -1.0f; }
    }
  }
  __syncthreads();
  for (int i = tid; i < KP1 * FF; i += 256) {        // gather xk
    int r = i >> 7, c = i & 127;
    sXk[i] = sB[sIdx[r] * FF + c];
  }
  for (int i = tid; i < KP1 * KP1; i += 256) {       // induced adj_k (raw adj)
    int r = i >> 6, c = i & 63;
    sAk[i] = (float)sT[sIdx[r] * NN + sIdx[c]];
  }
  __syncthreads();
  if (tid < KP1) {                                    // attention scores
    float f1 = 0.f, f2 = 0.f;
    for (int f = 0; f < FF; ++f) {
      float xv = (float)sXk[tid * FF + f];
      f1 += xv * att1[f]; f2 += xv * att1[FF + f];
    }
    sVec[tid] = f1; sVec[KP1 + tid] = f2;
  }
  __syncthreads();
  if (tid < KP1) {                                    // E = relu(f1_i+f2_j)+adj_k ; softmax rows
    float mx = -1e30f;
    for (int j = 0; j < KP1; ++j) {
      float e = fmaxf(sVec[tid] + sVec[KP1 + j], 0.f) + sAk[tid * KP1 + j];
      sAk[tid * KP1 + j] = e; mx = fmaxf(mx, e);
    }
    float ss = 0.f;
    for (int j = 0; j < KP1; ++j) { float e = expf(sAk[tid * KP1 + j] - mx); sAk[tid * KP1 + j] = e; ss += e; }
    float inv = 1.f / ss;
    for (int j = 0; j < KP1; ++j) sAk[tid * KP1 + j] *= inv;
  }
  if (tid < FF) {                                     // readout r1 (max|mean over K1 nodes)
    float mx = -1e30f, sm = 0.f;
    for (int i = 0; i < KP1; ++i) { float v = (float)sXk[i * FF + tid]; mx = fmaxf(mx, v); sm += v; }
    sGacc[tid]      += fmaxf(mx, 0.f);
    sGacc[FF + tid] += fmaxf(sm * (1.0f / KP1), 0.f);
  }
  __syncthreads();

  // ===== GCN layer 2 (64 nodes, adjacency a1) ===============================
  if (tid < KP1) {
    float s = 1.0f;
    for (int j = 0; j < KP1; ++j) s += sAk[tid * KP1 + j];
    sDinv[tid] = rsqrtf(fmaxf(s, 1e-12f));
  }
  __syncthreads();
  for (int i = tid; i < KP1 * KP1; i += 256) {        // An2 -> sT (ld=64)
    int r = i >> 6, c = i & 63;
    float av = sAk[i] + (r == c ? 1.0f : 0.0f);
    sT[i] = (_Float16)(sDinv[r] * av * sDinv[c]);
  }
  __syncthreads();
  mm_f16<KP1, FF, KP1, KP1, FF, FF, false, false>(sT, sXk, sA, nullptr, wave, lane); // T2 -> sA
  __syncthreads();
  for (int i = tid; i < FF * FF; i += 256) sT[i] = (_Float16)W2[i];
  __syncthreads();
  mm_f16<KP1, FF, FF, FF, FF, FF, true, true>(sA, sT, sB, b2, wave, lane);           // h2 -> sB
#pragma unroll
  for (int o = 0; o < FF * FF; o += 256 * 16)
    __builtin_prefetch(W3 + o + tid * 16, 0, 1);
  __syncthreads();

  // ===== Pool 2: info score on a1 ==========================================
  if (tid < KP1) {
    float s = 0.f;
    for (int j = 0; j < KP1; ++j) s += sAk[tid * KP1 + j];
    sDinv[tid]  = 1.0f / fmaxf(s, 1.0f);
    sScore[tid] = 0.f;
  }
  for (int i = tid; i < KP1 * KP1; i += 256) sA[i] = (_Float16)sAk[i];               // a1 f16 (ld=64)
  __syncthreads();
  mm_score<KP1, FF, KP1, KP1, FF>(sA, sB, sDinv, sScore, wave, lane);
  __syncthreads();
  if (wave == 0) {                                   // parallel top-K2
    for (int s = 0; s < KP2; ++s) {
      int bi = wave_argmax(sScore, KP1, lane);
      if (lane == 0) { sIdx[s] = bi; ((volatile float*)sScore)[bi] = -1.0f; }
    }
  }
  __syncthreads();
  for (int i = tid; i < KP2 * FF; i += 256) {
    int r = i >> 7, c = i & 127;
    sXk2[i] = sB[sIdx[r] * FF + c];
  }
  for (int i = tid; i < KP2 * KP2; i += 256) {
    int r = i >> 5, c = i & 31;
    sAk2[i] = sAk[sIdx[r] * KP1 + sIdx[c]];
  }
  __syncthreads();
  if (tid < KP2) {
    float f1 = 0.f, f2 = 0.f;
    for (int f = 0; f < FF; ++f) {
      float xv = (float)sXk2[tid * FF + f];
      f1 += xv * att2[f]; f2 += xv * att2[FF + f];
    }
    sVec[tid] = f1; sVec[KP2 + tid] = f2;
  }
  __syncthreads();
  if (tid < KP2) {
    float mx = -1e30f;
    for (int j = 0; j < KP2; ++j) {
      float e = fmaxf(sVec[tid] + sVec[KP2 + j], 0.f) + sAk2[tid * KP2 + j];
      sAk2[tid * KP2 + j] = e; mx = fmaxf(mx, e);
    }
    float ss = 0.f;
    for (int j = 0; j < KP2; ++j) { float e = expf(sAk2[tid * KP2 + j] - mx); sAk2[tid * KP2 + j] = e; ss += e; }
    float inv = 1.f / ss;
    for (int j = 0; j < KP2; ++j) sAk2[tid * KP2 + j] *= inv;
  }
  if (tid < FF) {                                     // readout r2
    float mx = -1e30f, sm = 0.f;
    for (int i = 0; i < KP2; ++i) { float v = (float)sXk2[i * FF + tid]; mx = fmaxf(mx, v); sm += v; }
    sGacc[tid]      += fmaxf(mx, 0.f);
    sGacc[FF + tid] += fmaxf(sm * (1.0f / KP2), 0.f);
  }
  __syncthreads();

  // ===== GCN layer 3 (32 nodes, adjacency a2) ===============================
  if (tid < KP2) {
    float s = 1.0f;
    for (int j = 0; j < KP2; ++j) s += sAk2[tid * KP2 + j];
    sDinv[tid] = rsqrtf(fmaxf(s, 1e-12f));
  }
  __syncthreads();
  for (int i = tid; i < KP2 * KP2; i += 256) {        // An3 -> sA (ld=32)
    int r = i >> 5, c = i & 31;
    float av = sAk2[i] + (r == c ? 1.0f : 0.0f);
    sA[i] = (_Float16)(sDinv[r] * av * sDinv[c]);
  }
  __syncthreads();
  mm_f16<KP2, FF, KP2, KP2, FF, FF, false, false>(sA, sXk2, sT, nullptr, wave, lane); // T3 -> sT
  __syncthreads();
  for (int i = tid; i < FF * FF; i += 256) sA[i] = (_Float16)W3[i];
  __syncthreads();
  mm_f16<KP2, FF, FF, FF, FF, FF, true, true>(sT, sA, sB, b3, wave, lane);            // h3 -> sB
  __syncthreads();
  if (tid < FF) {                                     // readout r3
    float mx = -1e30f, sm = 0.f;
    for (int i = 0; i < KP2; ++i) { float v = (float)sB[i * FF + tid]; mx = fmaxf(mx, v); sm += v; }
    sGacc[tid]      += fmaxf(mx, 0.f);
    sGacc[FF + tid] += fmaxf(sm * (1.0f / KP2), 0.f);
  }
  __syncthreads();

  // ===== MLP head + log_softmax ============================================
  if (tid < FF) {                                     // g1 = relu(g @ l1w + l1b)
    float s = l1b[tid];
    for (int i = 0; i < 2 * FF; ++i) s += sGacc[i] * l1w[i * FF + tid];
    sVec[tid] = fmaxf(s, 0.f);
  }
  __syncthreads();
  if (tid < 64) {                                     // g2 = relu(g1 @ l2w + l2b)
    float s = l2b[tid];
    for (int i = 0; i < FF; ++i) s += sVec[i] * l2w[i * 64 + tid];
    sVec[FF + tid] = fmaxf(s, 0.f);
  }
  __syncthreads();
  if (tid < 10) {                                     // logits
    float s = l3b[tid];
    for (int i = 0; i < 64; ++i) s += sVec[FF + i] * l3w[i * 10 + tid];
    sScore[tid] = s;
  }
  __syncthreads();
  if (tid == 0) {
    float mx = -1e30f;
    for (int j = 0; j < 10; ++j) mx = fmaxf(mx, sScore[j]);
    float ss = 0.f;
    for (int j = 0; j < 10; ++j) ss += expf(sScore[j] - mx);
    sScore[10] = mx + logf(ss);
  }
  __syncthreads();
  if (tid < 10) out[(size_t)bidx * 10 + tid] = sScore[tid] - sScore[10];
}

extern "C" void kernel_launch(void* const* d_in, const int* in_sizes, int n_in,
                              void* d_out, int out_size, void* d_ws, size_t ws_size,
                              hipStream_t stream) {
  (void)in_sizes; (void)n_in; (void)d_ws; (void)ws_size; (void)out_size;
  const float* x    = (const float*)d_in[0];
  const float* adj  = (const float*)d_in[1];
  const float* W1   = (const float*)d_in[2];
  const float* b1   = (const float*)d_in[3];
  const float* W2   = (const float*)d_in[4];
  const float* b2   = (const float*)d_in[5];
  const float* W3   = (const float*)d_in[6];
  const float* b3   = (const float*)d_in[7];
  const float* att1 = (const float*)d_in[8];
  const float* att2 = (const float*)d_in[9];
  const float* l1w  = (const float*)d_in[10];
  const float* l1b  = (const float*)d_in[11];
  const float* l2w  = (const float*)d_in[12];
  const float* l2b  = (const float*)d_in[13];
  const float* l3w  = (const float*)d_in[14];
  const float* l3b  = (const float*)d_in[15];
  float* out = (float*)d_out;

  hgpsl_fused_kernel<<<dim3(NB), dim3(256), 0, stream>>>(
      x, adj, W1, b1, W2, b2, W3, b3, att1, att2,
      l1w, l1b, l2w, l2b, l3w, l3b, out);
}
